// GATLinkPredictor_46059229282719
// MI455X (gfx1250) — matrix operand order repaired and checked by reading, not compiled, not run on hardware
//
#include <hip/hip_runtime.h>
#include <hip/hip_bf16.h>
#include <math.h>

// ---------------- problem constants (match reference) ----------------
#define NNODES 50000
#define NEDGES 800000
#define NLBL   100000
#define FIN    128
#define NH     4
#define NC     64
#define ND     256          // NH*NC
#define ETOT   (NEDGES + NNODES)
#define NEG_SLOPE 0.2f
#define GN     4            // n-tiles per wave in node GEMM (16x64 strip)

typedef __attribute__((ext_vector_type(16))) _Float16 v16h;
typedef __attribute__((ext_vector_type(8)))  _Float16 v8h;
typedef __attribute__((ext_vector_type(8)))  float    v8f;

union Uh { v16h v; v8h h[2]; };

static __device__ __forceinline__ v8f wmma_f16(v16h a, v16h b, v8f c) {
    return __builtin_amdgcn_wmma_f32_16x16x32_f16(false, a, false, b, (short)0, c,
                                                  false, false);
}

// order-preserving int encoding of float for atomicMax
static __device__ __forceinline__ int ordenc(float f) {
    int i = __float_as_int(f);
    return i < 0 ? (i ^ 0x7fffffff) : i;
}
static __device__ __forceinline__ float orddec(int i) {
    return __int_as_float(i < 0 ? (i ^ 0x7fffffff) : i);
}

// ---------------- small utility kernels ----------------
__global__ void k_f32_to_f16(const float* __restrict__ in, _Float16* __restrict__ out, int n) {
    int i = blockIdx.x * blockDim.x + threadIdx.x;
    if (i < n) out[i] = (_Float16)in[i];
}

// in[rows][cols] (row-major f32) -> out[cols][rows] (row-major f16)
__global__ void k_transpose_f16(const float* __restrict__ in, _Float16* __restrict__ out,
                                int rows, int cols) {
    int i = blockIdx.x * blockDim.x + threadIdx.x;
    if (i < rows * cols) {
        int r = i / cols, c = i % cols;
        out[c * rows + r] = (_Float16)in[i];
    }
}

__global__ void k_fill_i32(int* __restrict__ p, int v, int n) {
    int i = blockIdx.x * blockDim.x + threadIdx.x;
    if (i < n) p[i] = v;
}

__global__ void k_fill_f32(float* __restrict__ p, float v, int n) {
    int i = blockIdx.x * blockDim.x + threadIdx.x;
    if (i < n) p[i] = v;
}

// ---------------- WMMA GEMM: C[M x Nt] = A[M x K] * Bt[Nt x K]^T ----------------
// A f16 row-major (lda == K), Bt f16 N-major (each row = a column of B, ldb == K).
// One wave per 16x(16*GN) strip: A fragment reused across GN WMMAs per K-step.
// All fragments of a K-step are loaded into distinct registers before the WMMAs
// so the backend can clause the loads and use partial s_wait_loadcnt overlap.
__global__ __launch_bounds__(256)
void k_gemm_f16(const _Float16* __restrict__ A, const _Float16* __restrict__ Bt,
                float* __restrict__ C, _Float16* __restrict__ Ch,
                int M, int Nt, int K) {
    int wave = blockIdx.x * (blockDim.x >> 5) + (threadIdx.x >> 5);
    int lane = threadIdx.x & 31;
    int ngrp  = Nt / (16 * GN);
    int total = (M >> 4) * ngrp;
    if (wave >= total) return;           // wave-uniform: EXEC all-1s inside

    int mt = wave / ngrp;
    int g  = wave - mt * ngrp;
    int l15 = lane & 15;
    int hi  = lane >> 4;                 // 0 or 1

    const _Float16* arow = A + (size_t)(mt * 16 + l15) * K + hi * 8;
    const _Float16* brow[GN];
#pragma unroll
    for (int j = 0; j < GN; ++j)
        brow[j] = Bt + (size_t)((g * GN + j) * 16 + l15) * K + hi * 8;

    v8f acc[GN];
#pragma unroll
    for (int j = 0; j < GN; ++j)
#pragma unroll
        for (int i = 0; i < 8; ++i) acc[j][i] = 0.0f;

    for (int k0 = 0; k0 < K; k0 += 32) {
        Uh au, bu[GN];
        au.h[0] = *(const v8h*)(arow + k0);
        au.h[1] = *(const v8h*)(arow + k0 + 16);
#pragma unroll
        for (int j = 0; j < GN; ++j) {
            bu[j].h[0] = *(const v8h*)(brow[j] + k0);
            bu[j].h[1] = *(const v8h*)(brow[j] + k0 + 16);
        }
#pragma unroll
        for (int j = 0; j < GN; ++j)
            acc[j] = wmma_f16(au.v, bu[j].v, acc[j]);
    }

    int m0 = mt * 16 + hi * 8;
#pragma unroll
    for (int j = 0; j < GN; ++j) {
        int n = (g * GN + j) * 16 + l15;
#pragma unroll
        for (int r = 0; r < 8; ++r) {
            size_t idx = (size_t)(m0 + r) * Nt + n;
            float v = acc[j][r];
            C[idx] = v;
            Ch[idx] = (_Float16)v;
        }
    }
}

// ---------------- attention logits: al[n,h] = sum_c H[n,h,c]*a[h,c] ----------------
__global__ void k_logits(const float* __restrict__ H,
                         const float* __restrict__ a_src, const float* __restrict__ a_dst,
                         float* __restrict__ al_src, float* __restrict__ al_dst) {
    int t = blockIdx.x * blockDim.x + threadIdx.x;
    if (t >= NNODES * NH) return;
    int n = t >> 2, h = t & 3;
    const float* hp = H + (size_t)n * ND + h * NC;
    const float* as = a_src + h * NC;
    const float* ad = a_dst + h * NC;
    float s = 0.f, d = 0.f;
#pragma unroll 4
    for (int c = 0; c < NC; c += 4) {
        float4 hv = *(const float4*)(hp + c);
        float4 sv = *(const float4*)(as + c);
        float4 dv = *(const float4*)(ad + c);
        s += hv.x * sv.x + hv.y * sv.y + hv.z * sv.z + hv.w * sv.w;
        d += hv.x * dv.x + hv.y * dv.y + hv.z * dv.z + hv.w * dv.w;
    }
    al_src[t] = s;
    al_dst[t] = d;
}

static __device__ __forceinline__ void edge_sd(const int* __restrict__ ei, int e, int& s, int& d) {
    if (e < NEDGES) { s = ei[e]; d = ei[NEDGES + e]; }
    else            { s = d = e - NEDGES; }
}

// pass 1: per-edge leaky-relu score, atomic segment-max over dst
__global__ void k_edge_max(const int* __restrict__ ei,
                           const float* __restrict__ als, const float* __restrict__ ald,
                           int* __restrict__ mEnc) {
    int e = blockIdx.x * blockDim.x + threadIdx.x;
    if (e >= ETOT) return;
    int s, d; edge_sd(ei, e, s, d);
    float4 a = *(const float4*)(als + (size_t)s * NH);
    float4 b = *(const float4*)(ald + (size_t)d * NH);
    float v[4] = {a.x + b.x, a.y + b.y, a.z + b.z, a.w + b.w};
#pragma unroll
    for (int h = 0; h < NH; ++h) {
        float f = v[h] > 0.f ? v[h] : NEG_SLOPE * v[h];
        atomicMax(&mEnc[d * NH + h], ordenc(f));
    }
}

// pass 2: p = exp(e - max), cache p, atomic segment-sum of denom
__global__ void k_edge_sum(const int* __restrict__ ei,
                           const float* __restrict__ als, const float* __restrict__ ald,
                           const int* __restrict__ mEnc,
                           float* __restrict__ P, float* __restrict__ denom) {
    int e = blockIdx.x * blockDim.x + threadIdx.x;
    if (e >= ETOT) return;
    int s, d; edge_sd(ei, e, s, d);
    float4 a = *(const float4*)(als + (size_t)s * NH);
    float4 b = *(const float4*)(ald + (size_t)d * NH);
    float v[4] = {a.x + b.x, a.y + b.y, a.z + b.z, a.w + b.w};
#pragma unroll
    for (int h = 0; h < NH; ++h) {
        float f = v[h] > 0.f ? v[h] : NEG_SLOPE * v[h];
        float p = __expf(f - orddec(mEnc[d * NH + h]));
        P[(size_t)e * NH + h] = p;
        atomicAdd(&denom[d * NH + h], p);
    }
}

// pass 3: one wave per edge; lane covers 8 contiguous channels (head = lane/8)
__global__ __launch_bounds__(256)
void k_aggregate(const int* __restrict__ ei, const float* __restrict__ P,
                 const float* __restrict__ denom, const _Float16* __restrict__ Hh,
                 float* __restrict__ agg) {
    int e    = blockIdx.x * (blockDim.x >> 5) + (threadIdx.x >> 5);
    int lane = threadIdx.x & 31;
    if (e >= ETOT) return;
    int s, d; edge_sd(ei, e, s, d);
    int h = lane >> 3;
    float alpha = P[(size_t)e * NH + h] / denom[d * NH + h];
    const _Float16* hp = Hh + (size_t)s * ND + lane * 8;
    float*          op = agg + (size_t)d * ND + lane * 8;
    v8h hv = *(const v8h*)hp;
#pragma unroll
    for (int j = 0; j < 8; ++j) atomicAdd(op + j, (float)hv[j] * alpha);
}

// z = elu(agg + bias), stored f16 (feeds next GEMM / scorer gather)
__global__ void k_finalize(const float* __restrict__ agg, const float* __restrict__ bias,
                           _Float16* __restrict__ zh, int n) {
    int i = blockIdx.x * blockDim.x + threadIdx.x;
    if (i >= n) return;
    float v = agg[i] + bias[i & (ND - 1)];
    v = v > 0.f ? v : (__expf(v) - 1.0f);
    zh[i] = (_Float16)v;
}

// ---------------- fused link scorer ----------------
// one wave per tile of 16 label edges: gather -> 16x512x64 WMMA -> relu -> dot w/ Ws2
__global__ __launch_bounds__(32)
void k_score(const int* __restrict__ eli, const _Float16* __restrict__ Z,
             const _Float16* __restrict__ Ws1t, const float* __restrict__ bs1,
             const float* __restrict__ Ws2, const float* __restrict__ bs2,
             float* __restrict__ out) {
    __shared__ float hid[16 * 64];
    int lane = threadIdx.x;
    int tile = blockIdx.x;
    int m  = lane & 15;
    int hi = lane >> 4;

    int ea = eli[tile * 16 + m];
    int eb = eli[NLBL + tile * 16 + m];
    const _Float16* ra = Z + (size_t)ea * ND;
    const _Float16* rb = Z + (size_t)eb * ND;

    v8f acc[4];
#pragma unroll
    for (int t = 0; t < 4; ++t)
#pragma unroll
        for (int i = 0; i < 8; ++i) acc[t][i] = 0.0f;

    for (int k0 = 0; k0 < 2 * ND; k0 += 32) {
        const _Float16* base = (k0 < ND) ? (ra + k0) : (rb + (k0 - ND));
        Uh au, bu[4];
        au.h[0] = *(const v8h*)(base + hi * 8);
        au.h[1] = *(const v8h*)(base + 16 + hi * 8);
#pragma unroll
        for (int nt = 0; nt < 4; ++nt) {
            const _Float16* brow = Ws1t + (size_t)(nt * 16 + m) * (2 * ND) + k0 + hi * 8;
            bu[nt].h[0] = *(const v8h*)(brow);
            bu[nt].h[1] = *(const v8h*)(brow + 16);
        }
#pragma unroll
        for (int nt = 0; nt < 4; ++nt)
            acc[nt] = wmma_f16(au.v, bu[nt].v, acc[nt]);
    }

#pragma unroll
    for (int nt = 0; nt < 4; ++nt) {
        int n = nt * 16 + m;
        float bv = bs1[n];
#pragma unroll
        for (int r = 0; r < 8; ++r) {
            float v = acc[nt][r] + bv;
            hid[(8 * hi + r) * 64 + n] = v > 0.f ? v : 0.f;
        }
    }
    __syncthreads();
    if (lane < 16) {
        float sum = bs2[0];
#pragma unroll 8
        for (int c = 0; c < 64; ++c) sum += hid[lane * 64 + c] * Ws2[c];
        out[tile * 16 + lane] = sum;
    }
}

// ---------------- host launcher ----------------
static inline int cdiv(int a, int b) { return (a + b - 1) / b; }

extern "C" void kernel_launch(void* const* d_in, const int* in_sizes, int n_in,
                              void* d_out, int out_size, void* d_ws, size_t ws_size,
                              hipStream_t stream) {
    const float* x      = (const float*)d_in[0];
    const int*   ei     = (const int*)d_in[1];
    const int*   eli    = (const int*)d_in[2];
    const float* W1     = (const float*)d_in[3];
    const float* a_src1 = (const float*)d_in[4];
    const float* a_dst1 = (const float*)d_in[5];
    const float* b1     = (const float*)d_in[6];
    const float* W2     = (const float*)d_in[7];
    const float* a_src2 = (const float*)d_in[8];
    const float* a_dst2 = (const float*)d_in[9];
    const float* b2     = (const float*)d_in[10];
    const float* Ws1    = (const float*)d_in[11];
    const float* bs1    = (const float*)d_in[12];
    const float* Ws2    = (const float*)d_in[13];
    const float* bs2    = (const float*)d_in[14];
    float* out = (float*)d_out;

    // ---- carve workspace ----
    char* w = (char*)d_ws;
    auto carve = [&](size_t bytes) {
        void* p = (void*)w;
        w += (bytes + 255) & ~(size_t)255;
        return p;
    };
    _Float16* xh   = (_Float16*)carve((size_t)NNODES * FIN * 2);
    _Float16* w1t  = (_Float16*)carve((size_t)ND * FIN * 2);
    _Float16* w2t  = (_Float16*)carve((size_t)ND * ND * 2);
    _Float16* ws1t = (_Float16*)carve((size_t)NC * (2 * ND) * 2);
    float*    H    = (float*)   carve((size_t)NNODES * ND * 4);  // f32 features; reused as agg
    _Float16* Hh   = (_Float16*)carve((size_t)NNODES * ND * 2);  // f16 features (edge gather)
    _Float16* zh   = (_Float16*)carve((size_t)NNODES * ND * 2);  // elu output f16
    float*    als  = (float*)   carve((size_t)NNODES * NH * 4);
    float*    ald  = (float*)   carve((size_t)NNODES * NH * 4);
    int*      mEnc = (int*)     carve((size_t)NNODES * NH * 4);
    float*    den  = (float*)   carve((size_t)NNODES * NH * 4);
    float*    P    = (float*)   carve((size_t)ETOT * NH * 4);
    float*    agg  = H;  // alias: H is dead after k_logits in each layer

    const int T = 256;

    // ---- precision conversion + weight transposes ----
    k_f32_to_f16<<<cdiv(NNODES * FIN, T), T, 0, stream>>>(x, xh, NNODES * FIN);
    k_transpose_f16<<<cdiv(FIN * ND, T), T, 0, stream>>>(W1, w1t, FIN, ND);
    k_transpose_f16<<<cdiv(ND * ND, T), T, 0, stream>>>(W2, w2t, ND, ND);
    k_transpose_f16<<<cdiv(2 * ND * NC, T), T, 0, stream>>>(Ws1, ws1t, 2 * ND, NC);

    // ---- two GAT layers ----
    for (int layer = 0; layer < 2; ++layer) {
        const _Float16* A  = (layer == 0) ? xh  : zh;
        const _Float16* Bt = (layer == 0) ? w1t : w2t;
        const int K        = (layer == 0) ? FIN : ND;
        const float* asv   = (layer == 0) ? a_src1 : a_src2;
        const float* adv   = (layer == 0) ? a_dst1 : a_dst2;
        const float* bias  = (layer == 0) ? b1 : b2;

        int gemm_waves = (NNODES / 16) * (ND / (16 * GN));
        k_gemm_f16<<<cdiv(gemm_waves, 8), 256, 0, stream>>>(A, Bt, H, Hh, NNODES, ND, K);

        k_logits<<<cdiv(NNODES * NH, T), T, 0, stream>>>(H, asv, adv, als, ald);

        // H consumed -> reuse as agg accumulator
        k_fill_i32<<<cdiv(NNODES * NH, T), T, 0, stream>>>(mEnc, (int)0x80000000, NNODES * NH);
        k_fill_f32<<<cdiv(NNODES * NH, T), T, 0, stream>>>(den, 0.0f, NNODES * NH);
        k_fill_f32<<<cdiv(NNODES * ND, T), T, 0, stream>>>(agg, 0.0f, NNODES * ND);

        k_edge_max<<<cdiv(ETOT, T), T, 0, stream>>>(ei, als, ald, mEnc);
        k_edge_sum<<<cdiv(ETOT, T), T, 0, stream>>>(ei, als, ald, mEnc, P, den);
        k_aggregate<<<cdiv(ETOT, 8), 256, 0, stream>>>(ei, P, den, Hh, agg);

        k_finalize<<<cdiv(NNODES * ND, T), T, 0, stream>>>(agg, bias, zh, NNODES * ND);
    }

    // ---- fused link scorer over 100k label edges ----
    k_score<<<NLBL / 16, 32, 0, stream>>>(eli, zh, ws1t, bs1, Ws2, bs2, out);
}
